// Attention_79413945303230
// MI455X (gfx1250) — matrix operand order repaired and checked by reading, not compiled, not run on hardware
//
#include <hip/hip_runtime.h>
#include <cstdint>
#include <cstddef>

// ---------------------------------------------------------------------------
// Problem constants (from reference): B=2, T=2048, C=512, NB=4
// ---------------------------------------------------------------------------
constexpr int Bc  = 2;
constexpr int Tc  = 2048;
constexpr int Cc  = 512;
constexpr int NBc = 4;
constexpr int Mrows = Bc * Tc;          // 4096 rows for the projection GEMMs
constexpr float EPSc = 1.1920929e-07f;
constexpr float RSQRT_C = 0.04419417382415922f;   // 1/sqrt(512)
constexpr float LOG1E4  = 9.210340371976184f;     // ln(10000)

typedef __bf16 bf16_t;
typedef __attribute__((ext_vector_type(16))) __bf16 v16bf;
typedef __attribute__((ext_vector_type(8)))  __bf16 v8bf;
typedef __attribute__((ext_vector_type(8)))  float  v8f;

union F16x { v16bf v; v8bf h[2]; bf16_t e[16]; };
union F8x  { v8f v; float e[8]; };

__device__ __forceinline__ float softplusf(float x) {
    return (x > 20.0f) ? x : log1pf(__expf(x));
}

// ---------------------------------------------------------------------------
// Tensor Data Mover: one-descriptor 2D tile load global->LDS (ISA ch.8).
// D# group0: count=1, lds_addr, global_addr[56:0], type=2 ("image").
// D# group1: data_size=2B, tensor_dim0/1, tile_dim0/1, tensor_dim0_stride.
// Groups 2/3 zero (2D tensor). Tracked by TENSORcnt.
// ---------------------------------------------------------------------------
#if __has_builtin(__builtin_amdgcn_tensor_load_to_lds)
#define HAVE_TDM 1
typedef unsigned int tdm_u32x4 __attribute__((ext_vector_type(4)));
typedef int          tdm_i32x4 __attribute__((ext_vector_type(4)));
typedef int          tdm_i32x8 __attribute__((ext_vector_type(8)));

__device__ __forceinline__ void tdm_load_tile_2d(
        unsigned lds_addr, unsigned long long gaddr,
        unsigned tile_w, unsigned tile_h,
        unsigned tensor_w, unsigned tensor_h,
        unsigned long long row_stride_elems) {
    tdm_u32x4 g0;
    tdm_i32x8 g1;
    tdm_i32x4 gz4 = {0, 0, 0, 0};
    g0[0] = 1u;                                   // descriptor count = 1
    g0[1] = lds_addr;                             // lds_addr[31:0]
    g0[2] = (unsigned)gaddr;                      // global_addr[31:0]
    g0[3] = (unsigned)((gaddr >> 32) & 0x01FFFFFFull) | (2u << 30); // type=2
    g1[0] = (int)(1u << 16);                      // data_size=1 (2 bytes)
    g1[1] = (int)((tensor_w & 0xFFFFu) << 16);    // tensor_dim0[15:0] @bit48
    g1[2] = (int)(((tensor_w >> 16) & 0xFFFFu) |  // tensor_dim0[31:16]
                  ((tensor_h & 0xFFFFu) << 16));  // tensor_dim1[15:0]
    g1[3] = (int)(((tensor_h >> 16) & 0xFFFFu) |  // tensor_dim1[31:16]
                  ((tile_w & 0xFFFFu) << 16));    // tile_dim0
    g1[4] = (int)(tile_h & 0xFFFFu);              // tile_dim1 (tile_dim2=0)
    g1[5] = (int)(unsigned)(row_stride_elems & 0xFFFFFFFFull);
    g1[6] = (int)(unsigned)((row_stride_elems >> 32) & 0xFFFFull);
    g1[7] = 0;
#if __clang_major__ >= 23
    tdm_i32x8 gz8 = {0, 0, 0, 0, 0, 0, 0, 0};
    __builtin_amdgcn_tensor_load_to_lds(g0, g1, gz4, gz4, gz8, 0);
#else
    __builtin_amdgcn_tensor_load_to_lds(g0, g1, gz4, gz4, 0);
#endif
}
#endif

// Async global->LDS 16B transfer (ASYNCcnt-tracked, no VGPR round trip).
__device__ __forceinline__ void async_b128_to_lds(unsigned lds_off,
                                                  const void* gptr) {
    asm volatile("global_load_async_to_lds_b128 %0, %1, off"
                 :: "v"(lds_off), "v"(gptr) : "memory");
}
__device__ __forceinline__ void wait_asynccnt0() {
    asm volatile("s_wait_asynccnt 0x0" ::: "memory");
}

// ---------------------------------------------------------------------------
// f32 -> bf16 conversion (vectorized by 4)
// ---------------------------------------------------------------------------
__global__ __launch_bounds__(256) void k_cvt_bf16(const float* __restrict__ s,
                                                  bf16_t* __restrict__ d, int n) {
    int i = (blockIdx.x * 256 + threadIdx.x) * 4;
    if (i < n) {
        float4 v = *(const float4*)(s + i);
        d[i + 0] = (bf16_t)v.x;
        d[i + 1] = (bf16_t)v.y;
        d[i + 2] = (bf16_t)v.z;
        d[i + 3] = (bf16_t)v.w;
    }
}

// ---------------------------------------------------------------------------
// Generic bf16 WMMA GEMM: C[M,N](f32 or bf16) = A[M,K]bf16 @ B[K,N]bf16
// Tile: BM=128, BN=64, BK=32; 256 threads = 8 waves (wave32):
//   wave w -> mi = w>>1 (32-row strip), ni = w&1 (32-col strip); 2x2 frags.
// A tile staged via TDM tensor_load_to_lds (wave 0 issues one descriptor);
// B tile staged via per-thread global_load_async_to_lds_b128.
// ---------------------------------------------------------------------------
#define GBM 128
#define GBN 64
#define GBK 32

template <bool OUT_BF16>
__global__ __launch_bounds__(256)
void k_gemm_bf16(const bf16_t* __restrict__ A, const bf16_t* __restrict__ Bm,
                 float* __restrict__ Cf, bf16_t* __restrict__ Cb,
                 int M, int N, int K) {
    __shared__ bf16_t sA[GBM * GBK];   // 8 KB
    __shared__ bf16_t sB[GBK * GBN];   // 4 KB
    const int tid  = threadIdx.x;
    const int lane = tid & 31, w = tid >> 5;
    const int lr = lane & 15, lh = lane >> 4;
    const int mi = w >> 1, ni = w & 1;
    const int m0 = blockIdx.y * GBM, n0 = blockIdx.x * GBN;

    v8f acc[2][2] = {};

#if defined(HAVE_TDM)
    const unsigned ldsA_base = (unsigned)(size_t)&sA[0];
    const unsigned long long gaA =
        (unsigned long long)(size_t)&A[(size_t)m0 * K];
#endif

    for (int k0 = 0; k0 < K; k0 += GBK) {
#if defined(HAVE_TDM)
        // A tile (128x32 bf16) in one TDM descriptor, issued by wave 0.
        if (w == 0) {
            tdm_load_tile_2d(ldsA_base, gaA + (unsigned long long)k0 * 2,
                             GBK, GBM, (unsigned)(K - k0), (unsigned)GBM,
                             (unsigned long long)K);
        }
#else
        // Fallback: per-thread staging of the A tile.
#pragma unroll
        for (int i = 0; i < 2; ++i) {
            int chunk = tid * 2 + i;
            int ar = chunk >> 2, ac = (chunk & 3) * 8;
            *(v8bf*)&sA[ar * GBK + ac] =
                *(const v8bf*)&A[(size_t)(m0 + ar) * K + k0 + ac];
        }
        if (k0 + GBK < K)
            __builtin_prefetch(&A[(size_t)(m0 + (tid >> 1)) * K + k0 + GBK], 0, 1);
#endif
        // B tile (32x64 bf16): one async 16B global->LDS chunk per thread.
        {
            int br = tid >> 3, bc = (tid & 7) * 8;
            async_b128_to_lds((unsigned)(size_t)&sB[br * GBN + bc],
                              &Bm[(size_t)(k0 + br) * N + n0 + bc]);
        }
        wait_asynccnt0();
#if defined(HAVE_TDM)
        if (w == 0) __builtin_amdgcn_s_wait_tensorcnt(0);
#endif
        __syncthreads();

        F16x af[2], bfm[2];
#pragma unroll
        for (int f = 0; f < 2; ++f) {
            // A fragment: lane holds row M=lr, K = 8*lh..(+8) and 16+8*lh..(+8)
            const bf16_t* pa = &sA[(mi * 32 + f * 16 + lr) * GBK];
            af[f].h[0] = *(const v8bf*)(pa + 8 * lh);
            af[f].h[1] = *(const v8bf*)(pa + 16 + 8 * lh);
            // B fragment: lane holds col N=lr, K = 8*lh..(+8) and 16+8*lh..(+8)
            const bf16_t* pb = &sB[ni * 32 + f * 16 + lr];
            for (int e = 0; e < 8; ++e) {
                bfm[f].e[e]     = pb[(8 * lh + e) * GBN];
                bfm[f].e[8 + e] = pb[(16 + 8 * lh + e) * GBN];
            }
        }
#pragma unroll
        for (int fm = 0; fm < 2; ++fm)
#pragma unroll
            for (int fn = 0; fn < 2; ++fn)
                acc[fm][fn] = __builtin_amdgcn_wmma_f32_16x16x32_bf16(
                    false, af[fm].v, false, bfm[fn].v, (short)0, acc[fm][fn],
                    false, false);
        __syncthreads();
    }

    // Epilogue: D frag layout: VGPR r -> M = r + 8*lh, N = lr
#pragma unroll
    for (int fm = 0; fm < 2; ++fm)
#pragma unroll
        for (int fn = 0; fn < 2; ++fn) {
            F8x t; t.v = acc[fm][fn];
#pragma unroll
            for (int r = 0; r < 8; ++r) {
                int row = m0 + mi * 32 + fm * 16 + r + 8 * lh;
                int col = n0 + ni * 32 + fn * 16 + lr;
                if (OUT_BF16) Cb[(size_t)row * N + col] = (bf16_t)t.e[r];
                else          Cf[(size_t)row * N + col] = t.e[r];
            }
        }
}

// ---------------------------------------------------------------------------
// RMS-norm + RoPE on Q: reads Qtmp f32 (row-major [b*T+t, n*C+c]),
// writes q_bf bf16 laid out (B, NB, T, C). One 256-thread block per (b,t,n).
// ---------------------------------------------------------------------------
__global__ __launch_bounds__(256)
void k_rms_rope_q(const float* __restrict__ Qt, bf16_t* __restrict__ qb) {
    __shared__ float red[256];
    const int bid = blockIdx.x;            // (b*T+t)*NB + n
    const int n   = bid & (NBc - 1);
    const int row = bid >> 2;              // b*T + t
    const int t   = row & (Tc - 1);
    const int b   = row >> 11;
    const int i   = threadIdx.x;           // 0..255 (pairs i, i+256)

    const size_t src = (size_t)row * (NBc * Cc) + n * Cc;
    const float x1 = Qt[src + i];
    const float x2 = Qt[src + i + 256];
    red[i] = x1 * x1 + x2 * x2;
    __syncthreads();
    for (int off = 128; off; off >>= 1) {
        if (i < off) red[i] += red[i + off];
        __syncthreads();
    }
    const float rms = rsqrtf(red[0] * (1.0f / (float)Cc) + EPSc);

    const float inv = __expf(-(float)i * (LOG1E4 / 256.0f));
    float sn, cs;
    __sincosf((float)t * inv, &sn, &cs);
    const float a1 = x1 * rms, a2 = x2 * rms;

    const size_t dst = ((size_t)(b * NBc + n) * Tc + t) * Cc;
    qb[dst + i]       = (bf16_t)(a1 * cs - a2 * sn);
    qb[dst + i + 256] = (bf16_t)(a2 * cs + a1 * sn);
}

// ---------------------------------------------------------------------------
// RoPE on K, writing transposed kT (B, C, T) bf16 so the QK^T B-operand
// (lane = column s, elements = channels c) reads 32 contiguous bytes per channel over 16 lanes.
// ---------------------------------------------------------------------------
__global__ __launch_bounds__(256)
void k_rope_kT(const float* __restrict__ Kt, bf16_t* __restrict__ kT) {
    const int row = blockIdx.x;            // b*T + t
    const int t = row & (Tc - 1), b = row >> 11;
    const int i = threadIdx.x;
    const float x1 = Kt[(size_t)row * Cc + i];
    const float x2 = Kt[(size_t)row * Cc + i + 256];
    const float inv = __expf(-(float)i * (LOG1E4 / 256.0f));
    float sn, cs;
    __sincosf((float)t * inv, &sn, &cs);
    kT[((size_t)b * Cc + i) * Tc + t]       = (bf16_t)(x1 * cs - x2 * sn);
    kT[((size_t)b * Cc + i + 256) * Tc + t] = (bf16_t)(x2 * cs + x1 * sn);
}

// ---------------------------------------------------------------------------
// Attention scores pass: for each causal 64x64 (t,s) tile compute all 4
// branch scores via WMMA, then per element: max over branches, softplus
// weight u, 4-bit argmax winner mask. Writes u (bf16) and mask (u8),
// atomically accumulates row sums S and branch-activity bits.
// q/kT are L2-resident so fragments load straight from global.
// ---------------------------------------------------------------------------
__global__ __launch_bounds__(256)
void k_attn_scores(const bf16_t* __restrict__ qb, const bf16_t* __restrict__ kT,
                   bf16_t* __restrict__ ub, uint8_t* __restrict__ mb,
                   float* __restrict__ S, int* __restrict__ act) {
    const int st = blockIdx.x, tt = blockIdx.y, b = blockIdx.z;
    if (st > tt) return;                      // fully masked tile
    const int tid = threadIdx.x, lane = tid & 31, w = tid >> 5;
    const int lr = lane & 15, lh = lane >> 4;
    const int mi = w >> 1, sh = w & 1;
    const int tbase = tt * 64, sbase = st * 64;

    v8f acc[NBc][2] = {};
    const int trow = tbase + mi * 16 + lr;    // A-fragment row for this lane

    for (int k0 = 0; k0 < Cc; k0 += 32) {
        F16x af[NBc], bfm[2];
#pragma unroll
        for (int n = 0; n < NBc; ++n) {
            const bf16_t* pa = qb + (((size_t)(b * NBc + n) * Tc + trow) * Cc + k0);
            af[n].h[0] = *(const v8bf*)(pa + 8 * lh);
            af[n].h[1] = *(const v8bf*)(pa + 16 + 8 * lh);
        }
#pragma unroll
        for (int f = 0; f < 2; ++f) {
            const bf16_t* pb =
                kT + (((size_t)b * Cc + k0) * Tc + sbase + sh * 32 + f * 16 + lr);
            for (int e = 0; e < 8; ++e) {
                bfm[f].e[e]     = pb[(size_t)(8 * lh + e) * Tc];
                bfm[f].e[8 + e] = pb[(size_t)(16 + 8 * lh + e) * Tc];
            }
        }
#pragma unroll
        for (int n = 0; n < NBc; ++n)
#pragma unroll
            for (int f = 0; f < 2; ++f)
                acc[n][f] = __builtin_amdgcn_wmma_f32_16x16x32_bf16(
                    false, af[n].v, false, bfm[f].v, (short)0, acc[n][f],
                    false, false);
    }

    float rowsum[8];
    int   rowor[8];
#pragma unroll
    for (int r = 0; r < 8; ++r) { rowsum[r] = 0.0f; rowor[r] = 0; }

#pragma unroll
    for (int f = 0; f < 2; ++f) {
        F8x e0, e1, e2, e3;
        e0.v = acc[0][f]; e1.v = acc[1][f]; e2.v = acc[2][f]; e3.v = acc[3][f];
#pragma unroll
        for (int r = 0; r < 8; ++r) {
            const int gt = tbase + mi * 16 + r + 8 * lh;
            const int gs = sbase + sh * 32 + f * 16 + lr;
            const float a0 = e0.e[r] * RSQRT_C, a1 = e1.e[r] * RSQRT_C;
            const float a2 = e2.e[r] * RSQRT_C, a3 = e3.e[r] * RSQRT_C;
            const float mv = fmaxf(fmaxf(a0, a1), fmaxf(a2, a3));
            float u = 0.0f;
            int bits = 0;
            if (gs <= gt) {                    // unmasked (causal)
                u = softplusf(mv);
                bits = (a0 == mv) | ((a1 == mv) << 1) |
                       ((a2 == mv) << 2) | ((a3 == mv) << 3);
            }
            ub[(size_t)(b * Tc + gt) * Tc + gs] = (bf16_t)u;
            mb[(size_t)(b * Tc + gt) * Tc + gs] = (uint8_t)bits;
            rowsum[r] += u;
            rowor[r]  |= bits;
        }
    }
    // Cross-lane reduce within 16-lane halves (rows differ between halves).
#pragma unroll
    for (int r = 0; r < 8; ++r) {
        float s = rowsum[r];
        int   o = rowor[r];
        for (int m = 1; m < 16; m <<= 1) {
            s += __shfl_xor(s, m, 32);
            o |= __shfl_xor(o, m, 32);
        }
        if (lr == 0) {
            const int gt = tbase + mi * 16 + r + 8 * lh;
            atomicAdd(&S[b * Tc + gt], s);
            atomicOr(&act[b * Tc + gt], o);
        }
    }
}

// ---------------------------------------------------------------------------
// Routed PV pass: y(t,c) = scale(t) * sum_n sum_s (u(t,s)*win_n(t,s)) v_n(s,c)
//                 + sum_n activity_n * basis_n(c) + residual(t) * sink(c)
// 32 t-rows per block; 8 waves: mi = w>>2, cq = w&3 (128 cols, 8 N-frags).
// Masked bf16 A fragments rebuilt from u + winner mask; B-frags read the
// natural (B,T,NB*C) V layout directly. Output bf16 for the final GEMM.
// ---------------------------------------------------------------------------
__global__ __launch_bounds__(256)
void k_route_pv(const bf16_t* __restrict__ ub, const uint8_t* __restrict__ mb,
                const bf16_t* __restrict__ vb, const float* __restrict__ S,
                const int* __restrict__ act, const float* __restrict__ basis,
                const float* __restrict__ sres, bf16_t* __restrict__ yb) {
    const int tt = blockIdx.x, b = blockIdx.y;
    const int tid = threadIdx.x, lane = tid & 31, w = tid >> 5;
    const int lr = lane & 15, lh = lane >> 4;
    const int mi = w >> 2, cq = w & 3;
    const int t0 = tt * 32;
    const int trow = t0 + mi * 16 + lr;

    v8f acc[8] = {};
    const int nchunks = tt + 1;            // causal: s <= t

    for (int sc = 0; sc < nchunks; ++sc) {
        const int s0 = sc * 32;
        const bf16_t*  up = ub + ((size_t)(b * Tc + trow) * Tc + s0);
        const uint8_t* mp = mb + ((size_t)(b * Tc + trow) * Tc + s0);
        const v8bf u0 = *(const v8bf*)(up + 8 * lh);
        const v8bf u1 = *(const v8bf*)(up + 16 + 8 * lh);
        const unsigned long long q0 = *(const unsigned long long*)(mp + 8 * lh);
        const unsigned long long q1 = *(const unsigned long long*)(mp + 16 + 8 * lh);

        F16x af[NBc];
#pragma unroll
        for (int n = 0; n < NBc; ++n)
#pragma unroll
            for (int e = 0; e < 8; ++e) {
                af[n].e[e]     = ((q0 >> (8 * e)) & (1u << n)) ? u0[e] : (bf16_t)0.0f;
                af[n].e[8 + e] = ((q1 >> (8 * e)) & (1u << n)) ? u1[e] : (bf16_t)0.0f;
            }
#pragma unroll
        for (int n = 0; n < NBc; ++n) {
#pragma unroll
            for (int nf = 0; nf < 8; ++nf) {
                F16x bfm;
                const bf16_t* pb = vb + ((size_t)(b * Tc + s0) * (NBc * Cc) +
                                         n * Cc + cq * 128 + nf * 16 + lr);
                for (int e = 0; e < 8; ++e) {
                    bfm.e[e]     = pb[(size_t)(8 * lh + e) * (NBc * Cc)];
                    bfm.e[8 + e] = pb[(size_t)(16 + 8 * lh + e) * (NBc * Cc)];
                }
                acc[nf] = __builtin_amdgcn_wmma_f32_16x16x32_bf16(
                    false, af[n].v, false, bfm.v, (short)0, acc[nf], false, false);
            }
        }
    }

#pragma unroll
    for (int nf = 0; nf < 8; ++nf) {
        F8x e; e.v = acc[nf];
#pragma unroll
        for (int r = 0; r < 8; ++r) {
            const int gt = t0 + mi * 16 + r + 8 * lh;
            const int gc = cq * 128 + nf * 16 + lr;
            const float Sv    = S[b * Tc + gt];
            const float scale = fminf(1.0f / (Sv + 1e-6f), 1.0f);
            const float resid = 1.0f - Sv * scale;
            const int   ab    = act[b * Tc + gt];
            const bool  all   = (gt < Tc - 1);   // masked-column ties => all active
            float y = e.e[r] * scale;
#pragma unroll
            for (int n = 0; n < NBc; ++n)
                if (all || ((ab >> n) & 1)) y += basis[n * Cc + gc];
            y += resid * sres[gc];
            yb[(size_t)(b * Tc + gt) * Cc + gc] = (bf16_t)y;
        }
    }
}

// ---------------------------------------------------------------------------
// Host-side launcher
// ---------------------------------------------------------------------------
extern "C" void kernel_launch(void* const* d_in, const int* in_sizes, int n_in,
                              void* d_out, int out_size, void* d_ws, size_t ws_size,
                              hipStream_t stream) {
    const float* a   = (const float*)d_in[0];
    const float* x   = (const float*)d_in[1];
    const float* Wq  = (const float*)d_in[2];
    const float* Wk  = (const float*)d_in[3];
    const float* Wv  = (const float*)d_in[4];
    const float* Wo  = (const float*)d_in[5];
    const float* srs = (const float*)d_in[6];   // v_sink_residual (C)
    const float* sbs = (const float*)d_in[7];   // v_sink_basis (NB*C)

    char* wsb = (char*)d_ws;
    size_t off = 0;
    auto alloc = [&](size_t bytes) {
        size_t o = off;
        off += (bytes + 255) & ~(size_t)255;
        return o;
    };
    bf16_t* a_bf  = (bf16_t*)(wsb + alloc((size_t)Mrows * Cc * 2));
    bf16_t* x_bf  = (bf16_t*)(wsb + alloc((size_t)Mrows * Cc * 2));
    bf16_t* wq_bf = (bf16_t*)(wsb + alloc((size_t)Cc * NBc * Cc * 2));
    bf16_t* wk_bf = (bf16_t*)(wsb + alloc((size_t)Cc * Cc * 2));
    bf16_t* wv_bf = (bf16_t*)(wsb + alloc((size_t)Cc * NBc * Cc * 2));
    bf16_t* wo_bf = (bf16_t*)(wsb + alloc((size_t)Cc * Cc * 2));
    float*  qtmp  = (float*) (wsb + alloc((size_t)Mrows * NBc * Cc * 4));
    float*  ktmp  = (float*) (wsb + alloc((size_t)Mrows * Cc * 4));
    bf16_t* q_bf  = (bf16_t*)(wsb + alloc((size_t)Bc * NBc * Tc * Cc * 2));
    bf16_t* kT_bf = (bf16_t*)(wsb + alloc((size_t)Bc * Cc * Tc * 2));
    bf16_t* v_bf  = (bf16_t*)(wsb + alloc((size_t)Mrows * NBc * Cc * 2));
    bf16_t* u_bf  = (bf16_t*)(wsb + alloc((size_t)Bc * Tc * Tc * 2));
    uint8_t* m_u8 = (uint8_t*)(wsb + alloc((size_t)Bc * Tc * Tc));
    float*  Sbuf  = (float*) (wsb + alloc((size_t)Mrows * 4));
    int*    actb  = (int*)   (wsb + alloc((size_t)Mrows * 4));
    bf16_t* y_bf  = (bf16_t*)(wsb + alloc((size_t)Mrows * Cc * 2));
    (void)ws_size; (void)n_in; (void)in_sizes; (void)out_size;

    const dim3 blk(256);

    // 1) f32 -> bf16 converts (each n is a multiple of 1024)
    k_cvt_bf16<<<dim3(Mrows * Cc / 1024), blk, 0, stream>>>(a, a_bf, Mrows * Cc);
    k_cvt_bf16<<<dim3(Mrows * Cc / 1024), blk, 0, stream>>>(x, x_bf, Mrows * Cc);
    k_cvt_bf16<<<dim3(Cc * NBc * Cc / 1024), blk, 0, stream>>>(Wq, wq_bf, Cc * NBc * Cc);
    k_cvt_bf16<<<dim3(Cc * Cc / 1024), blk, 0, stream>>>(Wk, wk_bf, Cc * Cc);
    k_cvt_bf16<<<dim3(Cc * NBc * Cc / 1024), blk, 0, stream>>>(Wv, wv_bf, Cc * NBc * Cc);
    k_cvt_bf16<<<dim3(Cc * Cc / 1024), blk, 0, stream>>>(Wo, wo_bf, Cc * Cc);

    // 2) Projection GEMMs (bf16 WMMA, fp32 accumulate; TDM/async staging)
    k_gemm_bf16<false><<<dim3((NBc * Cc) / GBN, Mrows / GBM), blk, 0, stream>>>(
        a_bf, wq_bf, qtmp, nullptr, Mrows, NBc * Cc, Cc);
    k_gemm_bf16<true><<<dim3((NBc * Cc) / GBN, Mrows / GBM), blk, 0, stream>>>(
        a_bf, wv_bf, nullptr, v_bf, Mrows, NBc * Cc, Cc);
    k_gemm_bf16<false><<<dim3(Cc / GBN, Mrows / GBM), blk, 0, stream>>>(
        x_bf, wk_bf, ktmp, nullptr, Mrows, Cc, Cc);

    // 3) RMS-norm + RoPE on Q; RoPE + transpose on K
    k_rms_rope_q<<<dim3(Mrows * NBc), blk, 0, stream>>>(qtmp, q_bf);
    k_rope_kT<<<dim3(Mrows), blk, 0, stream>>>(ktmp, kT_bf);

    // 4) Zero the row-stat accumulators (re-done every call: deterministic)
    hipMemsetAsync(Sbuf, 0, (size_t)Mrows * 4, stream);
    hipMemsetAsync(actb, 0, (size_t)Mrows * 4, stream);

    // 5) Branch scores + routing weights + row stats
    k_attn_scores<<<dim3(Tc / 64, Tc / 64, Bc), blk, 0, stream>>>(
        q_bf, kT_bf, u_bf, m_u8, Sbuf, actb);

    // 6) Routed PV + fused scale/basis/residual epilogue -> y bf16
    k_route_pv<<<dim3(Tc / 32, Bc), blk, 0, stream>>>(
        u_bf, m_u8, v_bf, Sbuf, actb, sbs, srs, y_bf);

    // 7) Final projection y @ Wo -> d_out (f32)
    k_gemm_bf16<false><<<dim3(Cc / GBN, Mrows / GBM), blk, 0, stream>>>(
        y_bf, wo_bf, (float*)d_out, nullptr, Mrows, Cc, Cc);
}
